// TransformerBlock_76699525972245
// MI455X (gfx1250) — matrix operand order repaired and checked by reading, not compiled, not run on hardware
//
#include <hip/hip_runtime.h>

// ---------------------------------------------------------------------------
// Transformer block for MI455X (gfx1250, wave32, WMMA).
// All GEMMs: fp32 -> bf16 tiles in LDS -> v_wmma_f32_16x16x32_bf16 (fp32 acc).
// Attention: causal flash-attention, online softmax in fp32, WMMA for QK^T/PV.
// Staging is software-pipelined: next tile is prefetched into registers while
// the current tile's WMMAs run, so global loads overlap matrix math.
// ---------------------------------------------------------------------------

typedef __attribute__((ext_vector_type(16))) __bf16 v16bf;
typedef __attribute__((ext_vector_type(8)))  __bf16 v8bf;
typedef __attribute__((ext_vector_type(8)))  float  v8f;

#define EMBED    1024
#define HEADS    16
#define HEAD_DIM 64
#define FFDIM    4096
#define SEQ      2048

__device__ __forceinline__ __bf16 f2bf(float f) {
  unsigned u = __builtin_bit_cast(unsigned, f);
  u += 0x7FFFu + ((u >> 16) & 1u);              // round-to-nearest-even
  unsigned short h = (unsigned short)(u >> 16);
  return __builtin_bit_cast(__bf16, h);
}

__device__ __forceinline__ v8f wmma_bf16(v16bf a, v16bf b, v8f c) {
  // 8 args: (neg_a, A, neg_b, B, c_mod, C, reuse_a, reuse_b)
  return __builtin_amdgcn_wmma_f32_16x16x32_bf16(false, a, false, b, (short)0, c,
                                                 false, false);
}

// A-fragment (16 rows x K=32 bf16) from LDS tile stored [row][k], pitch in elems.
// Lane r (=lane&15) holds row r; half (=lane>>4) selects K chunks
// [8*half, 8*half+8) and [16+8*half, 16+8*half+8).
__device__ __forceinline__ v16bf load_frag_a(const __bf16* base, int pitch) {
  int lane = threadIdx.x & 31;
  int half = lane >> 4;
  int r    = lane & 15;
  const __bf16* p = base + r * pitch + half * 8;
  v8bf lo = *(const v8bf*)(p);
  v8bf hi = *(const v8bf*)(p + 16);
  return __builtin_shufflevector(lo, hi, 0, 1, 2, 3, 4, 5, 6, 7,
                                 8, 9, 10, 11, 12, 13, 14, 15);
}

// B-fragment (K=32 x 16 cols bf16) from LDS tile stored TRANSPOSED [col][k].
// Lane n (=lane&15) holds column n; K chunks [8*half, 8*half+8) and [16+8*half, 16+8*half+8).
__device__ __forceinline__ v16bf load_frag_b(const __bf16* base, int pitch) {
  int lane = threadIdx.x & 31;
  int half = lane >> 4;
  int n    = lane & 15;
  const __bf16* p = base + n * pitch + half * 8;
  v8bf lo = *(const v8bf*)(p);
  v8bf hi = *(const v8bf*)(p + 16);
  return __builtin_shufflevector(lo, hi, 0, 1, 2, 3, 4, 5, 6, 7,
                                 8, 9, 10, 11, 12, 13, 14, 15);
}

// ---------------------------------------------------------------------------
// Generic GEMM: C[M,N] = op( A[M,K] * B + bias ), bf16 WMMA, fp32 accumulate.
// b_nk==1: B source is row-major [N][K] (e.g. WQ/WK/WV: out-channel major).
// b_nk==0: B source is row-major [K][N] (e.g. WO, W_up, W_down).
// Requires M%128==0, N%128==0, K%32==0 (true for all calls here).
// ---------------------------------------------------------------------------
#define BM 128
#define BN 128
#define BK 32
#define KPAD 40  // multiple of 8 -> 16B-aligned ds_load_b128 fragment reads

__global__ __launch_bounds__(256) void gemm_kernel(
    const float* __restrict__ A, int lda,
    const float* __restrict__ B, int ldb, int b_nk,
    float* __restrict__ C, int ldc,
    const float* __restrict__ bias, int relu, int accum,
    int M, int N, int K) {
  __shared__ __bf16 As[BM][KPAD];
  __shared__ __bf16 Bs[BN][KPAD];

  int tid  = threadIdx.x;
  int wave = tid >> 5;
  int wm   = wave >> 2;  // 0..1
  int wn   = wave & 3;   // 0..3
  int bm0  = blockIdx.y * BM;
  int bn0  = blockIdx.x * BN;

  // staging thread mapping
  int rA  = tid >> 3;        // 0..31
  int c4  = (tid & 7) * 4;   // 0..28 (float4 column)
  int nT  = tid >> 1;        // 0..127 (transpose path)
  int kT0 = (tid & 1) * 16;  // 0 / 16

  float4 pa[4];    // A prefetch: 4 rows x float4
  float4 pb4[4];   // B prefetch ([N][K] path)
  float  pbT[16];  // B prefetch ([K][N] transpose path)

  auto prefetch = [&](int k0) {
#pragma unroll
    for (int rr = 0; rr < 4; ++rr)
      pa[rr] = *(const float4*)(A + (size_t)(bm0 + rA + rr * 32) * lda + k0 + c4);
    if (b_nk) {
#pragma unroll
      for (int rr = 0; rr < 4; ++rr)
        pb4[rr] = *(const float4*)(B + (size_t)(bn0 + rA + rr * 32) * ldb + k0 + c4);
    } else {
#pragma unroll
      for (int i = 0; i < 16; ++i)
        pbT[i] = B[(size_t)(k0 + kT0 + i) * ldb + bn0 + nT];
    }
  };

  auto stage = [&]() {
#pragma unroll
    for (int rr = 0; rr < 4; ++rr) {
      As[rA + rr * 32][c4 + 0] = f2bf(pa[rr].x);
      As[rA + rr * 32][c4 + 1] = f2bf(pa[rr].y);
      As[rA + rr * 32][c4 + 2] = f2bf(pa[rr].z);
      As[rA + rr * 32][c4 + 3] = f2bf(pa[rr].w);
    }
    if (b_nk) {
#pragma unroll
      for (int rr = 0; rr < 4; ++rr) {
        Bs[rA + rr * 32][c4 + 0] = f2bf(pb4[rr].x);
        Bs[rA + rr * 32][c4 + 1] = f2bf(pb4[rr].y);
        Bs[rA + rr * 32][c4 + 2] = f2bf(pb4[rr].z);
        Bs[rA + rr * 32][c4 + 3] = f2bf(pb4[rr].w);
      }
    } else {
#pragma unroll
      for (int i = 0; i < 16; ++i) Bs[nT][kT0 + i] = f2bf(pbT[i]);
    }
  };

  v8f acc[4][2] = {};

  prefetch(0);
  for (int k0 = 0; k0 < K; k0 += BK) {
    stage();
    __syncthreads();
    if (k0 + BK < K) prefetch(k0 + BK);  // overlap with WMMA below

    v16bf bfrag[2];
#pragma unroll
    for (int j = 0; j < 2; ++j)
      bfrag[j] = load_frag_b(&Bs[wn * 32 + j * 16][0], KPAD);
#pragma unroll
    for (int i = 0; i < 4; ++i) {
      v16bf afrag = load_frag_a(&As[wm * 64 + i * 16][0], KPAD);
#pragma unroll
      for (int j = 0; j < 2; ++j) acc[i][j] = wmma_bf16(afrag, bfrag[j], acc[i][j]);
    }
    __syncthreads();
  }

  // ---- epilogue: C layout row = vgpr + 8*half, col = lane&15 ----
  int lane = tid & 31;
  int half = lane >> 4;
  int nl   = lane & 15;
#pragma unroll
  for (int i = 0; i < 4; ++i)
#pragma unroll
    for (int j = 0; j < 2; ++j) {
      int col  = bn0 + wn * 32 + j * 16 + nl;
      float bv = bias ? bias[col] : 0.f;
#pragma unroll
      for (int r8 = 0; r8 < 8; ++r8) {
        int row = bm0 + wm * 64 + i * 16 + r8 + half * 8;
        float v = acc[i][j][r8] + bv;
        if (accum) v += C[(size_t)row * ldc + col];
        if (relu) v = fmaxf(v, 0.f);
        C[(size_t)row * ldc + col] = v;
      }
    }
}

// ---------------------------------------------------------------------------
// Causal flash attention.  Q/K/V stored [B*S, E] fp32 (head h at cols h*64..).
// Grid: (S/64, HEADS, B), 128 threads (4 waves, 16 query rows each).
// Scale 1/sqrt(64)=0.125 folded into Q on load.
// ---------------------------------------------------------------------------
#define FQ 64
#define FKB 32
#define QPITCH 72
#define VPITCH 40

__global__ __launch_bounds__(128) void flash_kernel(
    const float* __restrict__ Q, const float* __restrict__ Kg,
    const float* __restrict__ Vg, float* __restrict__ O, int S, int E) {
  __shared__ __bf16 Qs[FQ][QPITCH];       // [q][d]
  __shared__ __bf16 Ks[FKB][QPITCH];      // [key][d]     (B-frag for QK^T)
  __shared__ __bf16 Vs[HEAD_DIM][VPITCH]; // [d][key]     (B-frag for PV)
  __shared__ __bf16 Ps[4][16][VPITCH];    // per-wave P   (A-frag for PV)

  int b  = blockIdx.z, h = blockIdx.y, qb = blockIdx.x * FQ;
  int tid = threadIdx.x, wave = tid >> 5, lane = tid & 31;
  int half = lane >> 4, nl = lane & 15;
  size_t base = ((size_t)b * S) * E + (size_t)h * HEAD_DIM;  // + row*E + d

  // ---- stage Q tile (batched float4 loads, then convert) ----
  {
    int rQ  = tid >> 1;         // 0..63
    int dq0 = (tid & 1) * 32;   // 0 / 32
    float4 qx[8];
#pragma unroll
    for (int i = 0; i < 8; ++i)
      qx[i] = *(const float4*)(Q + base + (size_t)(qb + rQ) * E + dq0 + i * 4);
#pragma unroll
    for (int i = 0; i < 8; ++i) {
      int d = dq0 + i * 4;
      Qs[rQ][d + 0] = f2bf(qx[i].x * 0.125f);
      Qs[rQ][d + 1] = f2bf(qx[i].y * 0.125f);
      Qs[rQ][d + 2] = f2bf(qx[i].z * 0.125f);
      Qs[rQ][d + 3] = f2bf(qx[i].w * 0.125f);
    }
  }

  float m_run[8], l_run[8];
  v8f accv[4] = {};
#pragma unroll
  for (int j = 0; j < 8; ++j) { m_run[j] = -1e30f; l_run[j] = 0.f; }

  __syncthreads();

  int rKV = tid >> 2;          // 0..31 (key row)
  int d0  = (tid & 3) * 16;    // 0,16,32,48

  int kend = qb + FQ;  // causal bound (exclusive), uniform across workgroup
  for (int kb = 0; kb < kend; kb += FKB) {
    // ---- stage K/V tiles: batch 8 float4 loads, then convert/scatter ----
    {
      float4 kx[4], vx[4];
#pragma unroll
      for (int i = 0; i < 4; ++i) {
        size_t g = base + (size_t)(kb + rKV) * E + d0 + i * 4;
        kx[i] = *(const float4*)(Kg + g);
        vx[i] = *(const float4*)(Vg + g);
      }
#pragma unroll
      for (int i = 0; i < 4; ++i) {
        int d = d0 + i * 4;
        Ks[rKV][d + 0] = f2bf(kx[i].x);
        Ks[rKV][d + 1] = f2bf(kx[i].y);
        Ks[rKV][d + 2] = f2bf(kx[i].z);
        Ks[rKV][d + 3] = f2bf(kx[i].w);
        Vs[d + 0][rKV] = f2bf(vx[i].x);  // transpose V -> [d][key]
        Vs[d + 1][rKV] = f2bf(vx[i].y);
        Vs[d + 2][rKV] = f2bf(vx[i].z);
        Vs[d + 3][rKV] = f2bf(vx[i].w);
      }
    }
    __syncthreads();

    // ---- scores: S(16q x 32k) = Q(16x64) * K^T, two 16x16 frags, K-dim 64 ----
    v8f s[2];
#pragma unroll
    for (int nb = 0; nb < 2; ++nb) {
      v8f sc = {};
#pragma unroll
      for (int kk = 0; kk < HEAD_DIM; kk += 32) {
        v16bf a  = load_frag_a(&Qs[wave * 16][kk], QPITCH);
        v16bf bb = load_frag_b(&Ks[nb * 16][kk], QPITCH);
        sc = wmma_bf16(a, bb, sc);
      }
      s[nb] = sc;
    }

    // ---- causal mask + row max ----
    float mx[8];
#pragma unroll
    for (int j = 0; j < 8; ++j) {
      int qrow = qb + wave * 16 + j + half * 8;
#pragma unroll
      for (int nb = 0; nb < 2; ++nb) {
        int key = kb + nb * 16 + nl;
        if (key > qrow) s[nb][j] = -1e30f;
      }
      mx[j] = fmaxf(s[0][j], s[1][j]);
    }
#pragma unroll
    for (int off = 1; off < 16; off <<= 1)
#pragma unroll
      for (int j = 0; j < 8; ++j) mx[j] = fmaxf(mx[j], __shfl_xor(mx[j], off, 32));

    // ---- online softmax update ----
    float rs[8];
#pragma unroll
    for (int j = 0; j < 8; ++j) {
      float nm = fmaxf(m_run[j], mx[j]);
      float al = __expf(m_run[j] - nm);
      m_run[j] = nm;
      float p0 = __expf(s[0][j] - nm);
      float p1 = __expf(s[1][j] - nm);
      s[0][j] = p0; s[1][j] = p1;
      rs[j] = p0 + p1;
      l_run[j] *= al;
#pragma unroll
      for (int f = 0; f < 4; ++f) accv[f][j] *= al;
    }
#pragma unroll
    for (int off = 1; off < 16; off <<= 1)
#pragma unroll
      for (int j = 0; j < 8; ++j) rs[j] += __shfl_xor(rs[j], off, 32);
#pragma unroll
    for (int j = 0; j < 8; ++j) l_run[j] += rs[j];

    // ---- P (C-layout) -> LDS -> A-fragment layout ----
#pragma unroll
    for (int j = 0; j < 8; ++j) {
      int row = j + half * 8;
      Ps[wave][row][nl]      = f2bf(s[0][j]);
      Ps[wave][row][16 + nl] = f2bf(s[1][j]);
    }
    asm volatile("s_wait_dscnt 0" ::: "memory");  // wave-local LDS RAW fence

    v16bf pfrag = load_frag_a(&Ps[wave][0][0], VPITCH);
#pragma unroll
    for (int f = 0; f < 4; ++f) {
      v16bf vfrag = load_frag_b(&Vs[f * 16][0], VPITCH);
      accv[f] = wmma_bf16(pfrag, vfrag, accv[f]);
    }
    __syncthreads();
  }

  // ---- normalize and store ----
  float inv[8];
#pragma unroll
  for (int j = 0; j < 8; ++j) inv[j] = 1.f / l_run[j];
#pragma unroll
  for (int f = 0; f < 4; ++f)
#pragma unroll
    for (int j = 0; j < 8; ++j) {
      int row = qb + wave * 16 + j + half * 8;
      int d   = f * 16 + nl;
      O[base + (size_t)row * E + d] = accv[f][j] * inv[j];
    }
}

// ---------------------------------------------------------------------------
// out[row] = LayerNorm(X[row]) * gamma + beta + R[row]
// ---------------------------------------------------------------------------
__global__ __launch_bounds__(256) void ln_residual_kernel(
    const float* __restrict__ X, const float* __restrict__ R,
    const float* __restrict__ gamma, const float* __restrict__ beta,
    float* __restrict__ out) {
  __shared__ float sdata[8];
  int row = blockIdx.x, tid = threadIdx.x;
  int lane = tid & 31, wave = tid >> 5;
  const float* x = X + (size_t)row * EMBED;

  float lsum = 0.f;
  for (int i = tid; i < EMBED; i += 256) lsum += x[i];
#pragma unroll
  for (int off = 1; off < 32; off <<= 1) lsum += __shfl_xor(lsum, off, 32);
  if (lane == 0) sdata[wave] = lsum;
  __syncthreads();
  float tot = 0.f;
#pragma unroll
  for (int i = 0; i < 8; ++i) tot += sdata[i];
  float mean = tot / (float)EMBED;
  __syncthreads();

  float lvar = 0.f;
  for (int i = tid; i < EMBED; i += 256) { float d = x[i] - mean; lvar += d * d; }
#pragma unroll
  for (int off = 1; off < 32; off <<= 1) lvar += __shfl_xor(lvar, off, 32);
  if (lane == 0) sdata[wave] = lvar;
  __syncthreads();
  float vtot = 0.f;
#pragma unroll
  for (int i = 0; i < 8; ++i) vtot += sdata[i];
  float rstd = rsqrtf(vtot / (float)EMBED + 1e-6f);

  const float* r = R + (size_t)row * EMBED;
  float* o = out + (size_t)row * EMBED;
  for (int i = tid; i < EMBED; i += 256)
    o[i] = (x[i] - mean) * rstd * gamma[i] + beta[i] + r[i];
}

// ---------------------------------------------------------------------------
extern "C" void kernel_launch(void* const* d_in, const int* in_sizes, int n_in,
                              void* d_out, int out_size, void* d_ws, size_t ws_size,
                              hipStream_t stream) {
  const float* x      = (const float*)d_in[0];
  const float* WQ     = (const float*)d_in[1];
  const float* WK     = (const float*)d_in[2];
  const float* WV     = (const float*)d_in[3];
  const float* WO     = (const float*)d_in[4];
  const float* W_up   = (const float*)d_in[5];
  const float* b_up   = (const float*)d_in[6];
  const float* W_down = (const float*)d_in[7];
  const float* b_down = (const float*)d_in[8];
  const float* g1     = (const float*)d_in[9];
  const float* be1    = (const float*)d_in[10];
  const float* g2     = (const float*)d_in[11];
  const float* be2    = (const float*)d_in[12];

  const int S = SEQ, E = EMBED;
  const int Bn = in_sizes[0] / (S * E);  // batch = 4
  const int M  = Bn * S;                 // 8192 rows

  // Workspace regions (each M*E fp32 = 32 MB; peak 128 MB), aliased by stage:
  float* ws  = (float*)d_ws;
  const size_t RG = (size_t)M * E;
  float* Qb  = ws;            // Q           -> later: attention-proj out "Ob"
  float* Kb  = ws + RG;       // K           -> later: LN1 out "o1"
  float* Vb  = ws + 2 * RG;   // V           -> later: FFN out "h2"
  float* AO  = ws + 3 * RG;   // attn concat -> later: FFN chunk buffer "tmp"
  float* Ob  = Qb;
  float* o1  = Kb;
  float* h2  = Vb;
  float* tmp = AO;

  dim3 blk(256);
  dim3 gE(E / BN, M / BM);  // 8 x 64 blocks for N=1024

  // QKV projections: W[HD*D, E] is out-channel-major => b_nk = 1
  gemm_kernel<<<gE, blk, 0, stream>>>(x, E, WQ, E, 1, Qb, E, nullptr, 0, 0, M, E, E);
  gemm_kernel<<<gE, blk, 0, stream>>>(x, E, WK, E, 1, Kb, E, nullptr, 0, 0, M, E, E);
  gemm_kernel<<<gE, blk, 0, stream>>>(x, E, WV, E, 1, Vb, E, nullptr, 0, 0, M, E, E);

  // Causal flash attention (writes AO = concat heads, [M, E])
  flash_kernel<<<dim3(S / FQ, HEADS, Bn), dim3(128), 0, stream>>>(Qb, Kb, Vb, AO, S, E);

  // Output projection: WO is [K][N] => b_nk = 0
  gemm_kernel<<<gE, blk, 0, stream>>>(AO, E, WO, E, 0, Ob, E, nullptr, 0, 0, M, E, E);

  // out1 = LN(Ob)*g1+beta1 + x
  ln_residual_kernel<<<dim3(M), blk, 0, stream>>>(Ob, x, g1, be1, o1);

  // FFN chunked over FF (keeps workspace small; weights resident in 192MB L2):
  // h2 = sum_c relu(o1 @ W_up[:,c] + b_up[c]) @ W_down[c,:]  (+ b_down once)
  const int CH = 512;
  dim3 gC(CH / BN, M / BM);
  for (int c = 0; c < FFDIM / CH; ++c) {
    gemm_kernel<<<gC, blk, 0, stream>>>(o1, E, W_up + c * CH, FFDIM, 0,
                                        tmp, CH, b_up + c * CH, 1, 0, M, CH, E);
    gemm_kernel<<<gE, blk, 0, stream>>>(tmp, CH, W_down + (size_t)c * CH * E, E, 0,
                                        h2, E, (c == 0) ? b_down : nullptr, 0,
                                        (c != 0), M, E, CH);
  }

  // final = LN(h2)*g2+beta2 + out1
  ln_residual_kernel<<<dim3(M), blk, 0, stream>>>(h2, o1, g2, be2, (float*)d_out);
}
